// SeqLSTMAttnModel_7181185319153
// MI455X (gfx1250) — matrix operand order repaired and checked by reading, not compiled, not run on hardware
//
#include <hip/hip_runtime.h>
#include <hip/hip_bf16.h>

// ---------------------------------------------------------------------------
// Model dims
#define TT   128
#define BB   32
#define DIN  398
#define UN   256
#define G4U  1024
#define NSD  512
#define NCL  48

typedef __attribute__((ext_vector_type(16))) __bf16   v16bf;
typedef __attribute__((ext_vector_type(8)))  float    v8f;
typedef __attribute__((ext_vector_type(4)))  unsigned uint4v;

union U16BF {
  v16bf          v;
  uint4v         q[2];
  unsigned       u[8];
  unsigned short h[16];
};
union U8F {
  v8f    v;
  uint4v q[2];
  float  f[8];
};

__device__ __forceinline__ unsigned short f2bf(float f) {
  unsigned u = __builtin_bit_cast(unsigned, f);
  unsigned r = u + 0x7FFFu + ((u >> 16) & 1u);   // round-to-nearest-even
  return (unsigned short)(r >> 16);
}
__device__ __forceinline__ float bf2f(unsigned short h) {
  unsigned u = ((unsigned)h) << 16;
  return __builtin_bit_cast(float, u);
}
__device__ __forceinline__ float sigm(float x) { return 1.f / (1.f + __expf(-x)); }

__device__ __forceinline__ v8f wmma_bf16(v16bf a, v16bf b, v8f c) {
  return __builtin_amdgcn_wmma_f32_16x16x32_bf16(false, a, false, b, (short)0, c,
                                                 false, false);
}

// CDNA5 16-bit A-operand layout (16x32 tile):
//  lane = (m & 15) + 16*kh ; per lane the 16 bf16 are two contiguous runs:
//  K = 32*kt + 8*kh + [0..7]  and  K = 32*kt + 16 + 8*kh + [0..7]
__device__ __forceinline__ int a_kk(int j, int kh) {
  return (j < 4) ? (8 * kh + 2 * j) : (16 + 8 * kh + 2 * (j - 4));
}

// Build A operand from a row-major bf16 matrix in LDS with two b128 loads.
// Requires ldk (bf16 elements) to be a multiple of 8 -> rows 16B aligned.
__device__ __forceinline__ v16bf load_A_lds(const unsigned short* base, int mt,
                                            int kt, int lane, int ldk) {
  int m  = 16 * mt + (lane & 15);
  int kh = (lane >> 4) & 1;
  const unsigned short* row = base + m * ldk + 32 * kt + 8 * kh;
  U16BF a;
  a.q[0] = *(const uint4v*)(row);        // K run 0
  a.q[1] = *(const uint4v*)(row + 16);   // K run 1 (+16 elements)
  return a.v;
}

// Store one bf16 value into A-packed layout: [mt][Ktiles][lane(32)][16 halves]
__device__ __forceinline__ void store_hpack(unsigned short* Hpack, int Ktiles,
                                            int mrow, int k, unsigned short v) {
  int mt = mrow >> 4, la = mrow & 15;
  int kt = k >> 5, kk = k & 31;
  int kh, j;
  if (kk < 16) { kh = kk >> 3; j = (kk & 7) >> 1; }
  else         { int q = kk - 16; kh = q >> 3; j = 4 + ((q & 7) >> 1); }
  Hpack[(((mt * Ktiles + kt) * 32) + (la + 16 * kh)) * 16 + 2 * j + (kk & 1)] = v;
}

// ---------------------------------------------------------------------------
// Pack fp32 weight (K x N row-major) into WMMA B-operand tiles:
// dst u32 layout [nt][Ktiles][lane(32)][8]; lane half selects K 0..7,16..23 / 8..15,24..31.
__global__ void pack_b_kernel(const float* __restrict__ src, unsigned* __restrict__ dst,
                              int K, int N, int Ktiles, int Ntiles) {
  int idx   = blockIdx.x * blockDim.x + threadIdx.x;
  int total = Ntiles * Ktiles * 32 * 8;
  if (idx >= total) return;
  int j    = idx & 7;
  int lane = (idx >> 3) & 31;
  int rem  = idx >> 8;
  int kt   = rem % Ktiles;
  int nt   = rem / Ktiles;
  int n    = 16 * nt + (lane & 15);
  int kh   = lane >> 4;
  int k0   = 32 * kt + a_kk(j, kh);
  float f0 = (k0 < K && n < N) ? src[k0 * N + n] : 0.f;
  float f1 = (k0 + 1 < K && n < N) ? src[(k0 + 1) * N + n] : 0.f;
  dst[idx] = ((unsigned)f2bf(f1) << 16) | (unsigned)f2bf(f0);
}

// ---------------------------------------------------------------------------
// Embedding gather + concat directly into A-packed bf16 X: [mt(256)][kt(13)][32][8]
__device__ __forceinline__ float x_feature(int b, int t, int k,
                                           const int* words, const int* pos,
                                           const float* morph, const float* wemb,
                                           const float* pemb) {
  if (k < 300) return wemb[words[b * TT + t] * 300 + k];
  if (k < 332) return pemb[pos[b * TT + t] * 32 + (k - 300)];
  if (k < DIN) return morph[(b * TT + t) * 66 + (k - 332)];
  return 0.f;
}

__global__ void embed_pack_kernel(const int* __restrict__ words,
                                  const int* __restrict__ pos,
                                  const float* __restrict__ morph,
                                  const float* __restrict__ wemb,
                                  const float* __restrict__ pemb,
                                  unsigned* __restrict__ Xpack) {
  int idx   = blockIdx.x * blockDim.x + threadIdx.x;
  int total = 256 * 13 * 32 * 8;
  if (idx >= total) return;
  int j    = idx & 7;
  int lane = (idx >> 3) & 31;
  int rem  = idx >> 8;
  int kt   = rem % 13;
  int mt   = rem / 13;
  int m    = 16 * mt + (lane & 15);           // row = t*32 + b
  int t    = m >> 5, b = m & 31;
  int kh   = lane >> 4;
  int k0   = 32 * kt + a_kk(j, kh);
  float f0 = x_feature(b, t, k0, words, pos, morph, wemb, pemb);
  float f1 = x_feature(b, t, k0 + 1, words, pos, morph, wemb, pemb);
  Xpack[idx] = ((unsigned)f2bf(f1) << 16) | (unsigned)f2bf(f0);
}

// ---------------------------------------------------------------------------
// WMMA GEMM with 2x2 register blocking: each wave computes a 32x32 output
// block (2 M-tiles x 2 N-tiles), so each A/B fetch feeds two WMMAs.
// packedOut==1: store in accumulator layout [mt][Ntiles][lane][8] (b128 stores)
// packedOut==0: store row-major f32.
__global__ void __launch_bounds__(256) gemm_bias_kernel(
    const unsigned* __restrict__ Apack, const unsigned* __restrict__ Bpack,
    const float* __restrict__ bias, int Nbias, float* __restrict__ Z,
    int Mtiles, int Ntiles, int Ktiles, int packedOut) {
  int wave  = blockIdx.x * (blockDim.x >> 5) + (threadIdx.x >> 5);
  int lane  = threadIdx.x & 31;
  int nb2   = Ntiles >> 1;
  int total = (Mtiles >> 1) * nb2;
  if (wave >= total) return;
  int mb = wave / nb2, nb = wave % nb2;
  int mt0 = 2 * mb, nt0 = 2 * nb;
  v8f acc[2][2] = {};
  for (int kt = 0; kt < Ktiles; ++kt) {
    U16BF a0, a1, b0, b1;
    const uint4v* ap0 = (const uint4v*)(Apack + ((mt0 * Ktiles + kt) * 32 + lane) * 8);
    const uint4v* ap1 = (const uint4v*)(Apack + (((mt0 + 1) * Ktiles + kt) * 32 + lane) * 8);
    const uint4v* bp0 = (const uint4v*)(Bpack + ((nt0 * Ktiles + kt) * 32 + lane) * 8);
    const uint4v* bp1 = (const uint4v*)(Bpack + (((nt0 + 1) * Ktiles + kt) * 32 + lane) * 8);
    a0.q[0] = ap0[0]; a0.q[1] = ap0[1];
    a1.q[0] = ap1[0]; a1.q[1] = ap1[1];
    b0.q[0] = bp0[0]; b0.q[1] = bp0[1];
    b1.q[0] = bp1[0]; b1.q[1] = bp1[1];
    if (kt + 1 < Ktiles) {
      __builtin_prefetch(Bpack + ((nt0 * Ktiles + kt + 1) * 32 + lane) * 8, 0, 1);
      __builtin_prefetch(Apack + ((mt0 * Ktiles + kt + 1) * 32 + lane) * 8, 0, 1);
    }
    acc[0][0] = wmma_bf16(a0.v, b0.v, acc[0][0]);
    acc[0][1] = wmma_bf16(a0.v, b1.v, acc[0][1]);
    acc[1][0] = wmma_bf16(a1.v, b0.v, acc[1][0]);
    acc[1][1] = wmma_bf16(a1.v, b1.v, acc[1][1]);
  }
#pragma unroll
  for (int mi = 0; mi < 2; ++mi)
#pragma unroll
    for (int ni = 0; ni < 2; ++ni) {
      const int nt  = nt0 + ni;
      const int col = 16 * nt + (lane & 15);
      const float bv = (col < Nbias) ? bias[col] : 0.f;
      U8F o;
#pragma unroll
      for (int r = 0; r < 8; ++r) o.f[r] = acc[mi][ni][r] + bv;
      if (packedOut) {
        uint4v* zp = (uint4v*)(Z + (((mt0 + mi) * Ntiles + nt) * 32 + lane) * 8);
        zp[0] = o.q[0]; zp[1] = o.q[1];
      } else {
        const int N = Ntiles * 16;
#pragma unroll
        for (int r = 0; r < 8; ++r) {
          int row = 16 * (mt0 + mi) + r + 8 * (lane >> 4);
          Z[row * N + col] = o.f[r];
        }
      }
    }
}

// ---------------------------------------------------------------------------
// Sequential LSTM recurrence (U=256). Zx = x@Wi+b precomputed in packed
// accumulator layout [mt(256)][nt(64)][lane(32)][8] f32.
// One workgroup, 16 waves; wave w owns hidden cols [16w,16w+16) for all 4 gates.
__global__ void __launch_bounds__(512) lstm_seq_kernel(
    const float* __restrict__ Zx, const unsigned* __restrict__ Whpack,
    unsigned short* __restrict__ Hpack, float* __restrict__ Hf32) {
  __shared__ unsigned short hbf[32 * 256];
  const int tid = threadIdx.x, lane = tid & 31, w = tid >> 5;
  for (int i = tid; i < 32 * 256; i += 512) hbf[i] = 0;
  __syncthreads();
  v8f c[2] = {};
  for (int t = 0; t < TT; ++t) {
    v8f acc[4][2];
#pragma unroll
    for (int g2 = 0; g2 < 4; ++g2) {
      const int nt = w + 16 * g2;
#pragma unroll
      for (int mt = 0; mt < 2; ++mt) {
        const uint4v* zp =
            (const uint4v*)(Zx + (((2 * t + mt) * 64 + nt) * 32 + lane) * 8);
        U8F z;
        z.q[0] = zp[0]; z.q[1] = zp[1];
        acc[g2][mt] = z.v;
      }
    }
    for (int kt = 0; kt < 8; ++kt) {
      v16bf A0 = load_A_lds(hbf, 0, kt, lane, 256);
      v16bf A1 = load_A_lds(hbf, 1, kt, lane, 256);
#pragma unroll
      for (int g2 = 0; g2 < 4; ++g2) {
        const int nt = w + 16 * g2;
        U16BF bop;
        const uint4v* bp = (const uint4v*)(Whpack + ((nt * 8 + kt) * 32 + lane) * 8);
        bop.q[0] = bp[0]; bop.q[1] = bp[1];
        acc[g2][0] = wmma_bf16(A0, bop.v, acc[g2][0]);
        acc[g2][1] = wmma_bf16(A1, bop.v, acc[g2][1]);
      }
    }
    __syncthreads();                       // all reads of old h done
    const int u = 16 * w + (lane & 15);
#pragma unroll
    for (int mt = 0; mt < 2; ++mt)
#pragma unroll
      for (int r = 0; r < 8; ++r) {
        const float iv = sigm(acc[0][mt][r]);
        const float fv = sigm(acc[1][mt][r]);
        const float gv = tanhf(acc[2][mt][r]);
        const float ov = sigm(acc[3][mt][r]);
        float cv = fv * c[mt][r] + iv * gv;
        c[mt][r] = cv;
        const float hv = ov * tanhf(cv);
        const int b = 16 * mt + r + 8 * (lane >> 4);
        const unsigned short hb = f2bf(hv);
        hbf[b * 256 + u]             = hb;
        Hf32[(t * 32 + b) * 256 + u] = hv;
        store_hpack(Hpack, 8, t * 32 + b, u, hb);
      }
    __syncthreads();
  }
}

// ---------------------------------------------------------------------------
// Attention decoder: one workgroup, 32 waves, persistent over 128 steps.
__global__ void __launch_bounds__(1024) decoder_kernel(
    const float* __restrict__ aW1,   // 4096 x 64, = a@W1a + b1 (row-major)
    const float* __restrict__ Af32,  // (t*32+b) x 256
    const float* __restrict__ W1s,   // 512 x 50
    const float* __restrict__ W2,    // 50 x 10
    const float* __restrict__ b2, const float* __restrict__ W3,
    const float* __restrict__ b3,
    const unsigned* __restrict__ Wip,  // post_Wi packed [128][8][32][8]
    const unsigned* __restrict__ Whp,  // post_Wh packed [128][16][32][8]
    const float* __restrict__ pb,      // 2048
    const float* __restrict__ outW,    // 512 x 48
    const float* __restrict__ outb,    // 48
    float* __restrict__ out)           // (b*128+t)*48+cls
{
  extern __shared__ char smem[];
  unsigned short* sbf    = (unsigned short*)smem;                        // 32x512 bf16
  char*           mid    = smem + 32 * 512 * 2;                          // 16 KB region
  unsigned short* ctxbf  = (unsigned short*)mid;                         // 32x256 bf16
  float*          sW1    = (float*)mid;                                  // 32x52 (aliased)
  float*          logits = (float*)mid;                                  // 32x48 (aliased)
  float*          scores = (float*)(smem + 32 * 512 * 2 + 32 * 256 * 2); // 32x128
  const int tid = threadIdx.x, lane = tid & 31, w = tid >> 5;
  for (int i = tid; i < 32 * 512; i += 1024) sbf[i] = 0;
  __syncthreads();
  v8f c[2] = {};
  for (int t = 0; t < TT; ++t) {
    // (a) sW1 = s @ W1[256:,:]
    for (int idx = tid; idx < 32 * 50; idx += 1024) {
      int b = idx / 50, j = idx - b * 50;
      float a0 = 0.f;
      for (int k = 0; k < NSD; ++k) a0 += bf2f(sbf[b * NSD + k]) * W1s[k * 50 + j];
      sW1[b * 52 + j] = a0;
    }
    __syncthreads();
    // (b) attention MLP -> scores
    for (int idx = tid; idx < BB * TT; idx += 1024) {
      int b = idx & 31, tt = idx >> 5;
      const float* arow = aW1 + (tt * 32 + b) * 64;
      float e1[50];
      for (int j = 0; j < 50; ++j) e1[j] = tanhf(arow[j] + sW1[b * 52 + j]);
      float e3 = b3[0];
      for (int kk = 0; kk < 10; ++kk) {
        float a2 = b2[kk];
        for (int j = 0; j < 50; ++j) a2 += e1[j] * W2[j * 10 + kk];
        e3 += tanhf(a2) * W3[kk];
      }
      scores[b * TT + tt] = fmaxf(e3, 0.f);
    }
    __syncthreads();
    // (c) softmax over time
    if (tid < 32) {
      const int b = tid;
      float mx = -3.0e38f;
      for (int tt = 0; tt < TT; ++tt) mx = fmaxf(mx, scores[b * TT + tt]);
      float sm = 0.f;
      for (int tt = 0; tt < TT; ++tt) {
        float e = __expf(scores[b * TT + tt] - mx);
        scores[b * TT + tt] = e; sm += e;
      }
      float inv = 1.f / sm;
      for (int tt = 0; tt < TT; ++tt) scores[b * TT + tt] *= inv;
    }
    __syncthreads();
    // (d) ctx = sum_t alpha * a
    for (int idx = tid; idx < BB * 256; idx += 1024) {
      int b = idx >> 8, u = idx & 255;
      float a3 = 0.f;
      for (int tt = 0; tt < TT; ++tt)
        a3 += scores[b * TT + tt] * Af32[(tt * 32 + b) * 256 + u];
      ctxbf[b * 256 + u] = f2bf(a3);
    }
    __syncthreads();
    // (e) post-LSTM gates via WMMA: ctx@Wi (K=256) + s@Wh (K=512) + b
    v8f acc[4][2];
#pragma unroll
    for (int g2 = 0; g2 < 4; ++g2) {
      const float bb = pb[16 * (w + 32 * g2) + (lane & 15)];
#pragma unroll
      for (int mt = 0; mt < 2; ++mt)
#pragma unroll
        for (int r = 0; r < 8; ++r) acc[g2][mt][r] = bb;
    }
    for (int kt = 0; kt < 8; ++kt) {
      v16bf A0 = load_A_lds(ctxbf, 0, kt, lane, 256);
      v16bf A1 = load_A_lds(ctxbf, 1, kt, lane, 256);
#pragma unroll
      for (int g2 = 0; g2 < 4; ++g2) {
        const int nt = w + 32 * g2;
        U16BF bop;
        const uint4v* bp = (const uint4v*)(Wip + ((nt * 8 + kt) * 32 + lane) * 8);
        bop.q[0] = bp[0]; bop.q[1] = bp[1];
        acc[g2][0] = wmma_bf16(A0, bop.v, acc[g2][0]);
        acc[g2][1] = wmma_bf16(A1, bop.v, acc[g2][1]);
      }
    }
    for (int kt = 0; kt < 16; ++kt) {
      v16bf A0 = load_A_lds(sbf, 0, kt, lane, 512);
      v16bf A1 = load_A_lds(sbf, 1, kt, lane, 512);
#pragma unroll
      for (int g2 = 0; g2 < 4; ++g2) {
        const int nt = w + 32 * g2;
        U16BF bop;
        const uint4v* bp = (const uint4v*)(Whp + ((nt * 16 + kt) * 32 + lane) * 8);
        bop.q[0] = bp[0]; bop.q[1] = bp[1];
        acc[g2][0] = wmma_bf16(A0, bop.v, acc[g2][0]);
        acc[g2][1] = wmma_bf16(A1, bop.v, acc[g2][1]);
      }
    }
    __syncthreads();                 // all reads of ctx / old s complete
    // (f) cell update, publish new s
    const int u = 16 * w + (lane & 15);
#pragma unroll
    for (int mt = 0; mt < 2; ++mt)
#pragma unroll
      for (int r = 0; r < 8; ++r) {
        const float iv = sigm(acc[0][mt][r]);
        const float fv = sigm(acc[1][mt][r]);
        const float gv = tanhf(acc[2][mt][r]);
        const float ov = sigm(acc[3][mt][r]);
        float cv = fv * c[mt][r] + iv * gv;
        c[mt][r] = cv;
        const float sv = ov * tanhf(cv);
        const int b = 16 * mt + r + 8 * (lane >> 4);
        sbf[b * NSD + u] = f2bf(sv);
      }
    __syncthreads();
    // (g) output head logits
    for (int idx = tid; idx < BB * NCL; idx += 1024) {
      int b = idx / NCL, cls = idx - b * NCL;
      float a4 = outb[cls];
      for (int k = 0; k < NSD; ++k) a4 += bf2f(sbf[b * NSD + k]) * outW[k * NCL + cls];
      logits[b * NCL + cls] = a4;
    }
    __syncthreads();
    // (h) softmax over classes, write output
    if (tid < 32) {
      const int b = tid;
      float mx = -3.0e38f;
      for (int cls = 0; cls < NCL; ++cls) mx = fmaxf(mx, logits[b * NCL + cls]);
      float sm = 0.f;
      float ex[NCL];
      for (int cls = 0; cls < NCL; ++cls) {
        ex[cls] = __expf(logits[b * NCL + cls] - mx); sm += ex[cls];
      }
      float inv = 1.f / sm;
      for (int cls = 0; cls < NCL; ++cls)
        out[(b * TT + t) * NCL + cls] = ex[cls] * inv;
    }
    __syncthreads();
  }
}

// ---------------------------------------------------------------------------
extern "C" void kernel_launch(void* const* d_in, const int* in_sizes, int n_in,
                              void* d_out, int out_size, void* d_ws, size_t ws_size,
                              hipStream_t stream) {
  (void)in_sizes; (void)n_in; (void)out_size; (void)ws_size;
  const int*   words  = (const int*)d_in[0];
  const int*   pos    = (const int*)d_in[1];
  const float* morph  = (const float*)d_in[2];
  const float* wemb   = (const float*)d_in[3];
  const float* pemb   = (const float*)d_in[4];
  const float* l1Wi   = (const float*)d_in[5];
  const float* l1Wh   = (const float*)d_in[6];
  const float* l1b    = (const float*)d_in[7];
  const float* l2Wi   = (const float*)d_in[8];
  const float* l2Wh   = (const float*)d_in[9];
  const float* l2b    = (const float*)d_in[10];
  const float* aW1w   = (const float*)d_in[11];
  const float* ab1    = (const float*)d_in[12];
  const float* aW2    = (const float*)d_in[13];
  const float* ab2    = (const float*)d_in[14];
  const float* aW3    = (const float*)d_in[15];
  const float* ab3    = (const float*)d_in[16];
  const float* pWi    = (const float*)d_in[17];
  const float* pWh    = (const float*)d_in[18];
  const float* pb     = (const float*)d_in[19];
  const float* outW   = (const float*)d_in[20];
  const float* outb   = (const float*)d_in[21];
  float*       outp   = (float*)d_out;

  // workspace carving (256B aligned)
  size_t off = 0;
  auto carve = [&](size_t bytes) {
    void* p = (char*)d_ws + off;
    off += (bytes + 255) & ~(size_t)255;
    return p;
  };
  unsigned* Xpack     = (unsigned*)carve((size_t)256 * 13 * 256 * 4);
  unsigned* W1iP      = (unsigned*)carve((size_t)64 * 13 * 256 * 4);
  unsigned* W1hP      = (unsigned*)carve((size_t)64 * 8 * 256 * 4);
  unsigned* W2iP      = (unsigned*)carve((size_t)64 * 8 * 256 * 4);
  unsigned* W2hP      = (unsigned*)carve((size_t)64 * 8 * 256 * 4);
  unsigned* pWiP      = (unsigned*)carve((size_t)128 * 8 * 256 * 4);
  unsigned* pWhP      = (unsigned*)carve((size_t)128 * 16 * 256 * 4);
  unsigned* aW1aP     = (unsigned*)carve((size_t)4 * 8 * 256 * 4);
  float*    Zbuf      = (float*)carve((size_t)4096 * 1024 * 4);
  unsigned short* H1P = (unsigned short*)carve((size_t)256 * 8 * 32 * 16 * 2);
  unsigned short* H2P = (unsigned short*)carve((size_t)256 * 8 * 32 * 16 * 2);
  float*    Af32      = (float*)carve((size_t)4096 * 256 * 4);
  float*    aW1buf    = (float*)carve((size_t)4096 * 64 * 4);

  auto blk = [](int total) { return (total + 255) / 256; };

  // weight packing (bf16, WMMA B-operand layout)
  pack_b_kernel<<<blk(64 * 13 * 256), 256, 0, stream>>>(l1Wi, W1iP, 398, 1024, 13, 64);
  pack_b_kernel<<<blk(64 * 8 * 256), 256, 0, stream>>>(l1Wh, W1hP, 256, 1024, 8, 64);
  pack_b_kernel<<<blk(64 * 8 * 256), 256, 0, stream>>>(l2Wi, W2iP, 256, 1024, 8, 64);
  pack_b_kernel<<<blk(64 * 8 * 256), 256, 0, stream>>>(l2Wh, W2hP, 256, 1024, 8, 64);
  pack_b_kernel<<<blk(128 * 8 * 256), 256, 0, stream>>>(pWi, pWiP, 256, 2048, 8, 128);
  pack_b_kernel<<<blk(128 * 16 * 256), 256, 0, stream>>>(pWh, pWhP, 512, 2048, 16, 128);
  pack_b_kernel<<<blk(4 * 8 * 256), 256, 0, stream>>>(aW1w, aW1aP, 256, 50, 8, 4);

  // embeddings -> A-packed X
  embed_pack_kernel<<<blk(256 * 13 * 256), 256, 0, stream>>>(words, pos, morph,
                                                             wemb, pemb, Xpack);

  // Z1 = X @ l1Wi + b1   (packed-C layout, 128x32 = 4096 wave-blocks)
  gemm_bias_kernel<<<(128 * 32 + 7) / 8, 256, 0, stream>>>(Xpack, W1iP, l1b, 1024,
                                                           Zbuf, 256, 64, 13, 1);
  // LSTM-1 recurrence
  lstm_seq_kernel<<<1, 512, 0, stream>>>(Zbuf, W1hP, H1P, Af32);
  // Z2 = H1 @ l2Wi + b2   (packed-C layout)
  gemm_bias_kernel<<<(128 * 32 + 7) / 8, 256, 0, stream>>>((const unsigned*)H1P, W2iP,
                                                           l2b, 1024, Zbuf, 256, 64, 8, 1);
  // LSTM-2 recurrence -> a (Af32) + A-packed a (H2P)
  lstm_seq_kernel<<<1, 512, 0, stream>>>(Zbuf, W2hP, H2P, Af32);
  // aW1 = a @ attn_W1[:256] + b1   (row-major 4096 x 64, 128x2 wave-blocks)
  gemm_bias_kernel<<<(128 * 2 + 7) / 8, 256, 0, stream>>>((const unsigned*)H2P, aW1aP,
                                                          ab1, 50, aW1buf, 256, 4, 8, 0);
  // attention decoder (64 KB dynamic LDS)
  decoder_kernel<<<1, 1024, 65536, stream>>>(aW1buf, Af32, aW1w + 256 * 50, aW2, ab2,
                                             aW3, ab3, pWiP, pWhP, pb, outW, outb,
                                             outp);
}
